// TopologyAwareGATEncoder_10488310137492
// MI455X (gfx1250) — matrix operand order repaired and checked by reading, not compiled, not run on hardware
//
#include <hip/hip_runtime.h>
#include <hip/hip_bf16.h>
#include <stdint.h>

// ---------------------------------------------------------------------------
// TopologyAwareGATEncoder for MI455X (gfx1250), wave32 + WMMA f16->f32.
// N = 50000 (divisible by 16), E = 400000, HID = 64, HEADS = 4.
// GEMM A-panels staged to LDS via gfx1250 async global->LDS loads
// (ASYNCcnt path, global_load_async_to_lds_b128 + s_wait_asynccnt).
// ---------------------------------------------------------------------------

typedef __attribute__((ext_vector_type(16))) _Float16 v16h;
typedef __attribute__((ext_vector_type(8)))  _Float16 v8h;
typedef __attribute__((ext_vector_type(8)))  float    v8f;

#if defined(__HIP_DEVICE_COMPILE__) && defined(__has_builtin)
#  if __has_builtin(__builtin_amdgcn_global_load_async_to_lds_b128) && \
      __has_builtin(__builtin_amdgcn_s_wait_asynccnt)
#    define USE_ASYNC_LDS 1
#  endif
#endif
#ifndef USE_ASYNC_LDS
#  define USE_ASYNC_LDS 0
#endif

#if USE_ASYNC_LDS
typedef int v4i __attribute__((ext_vector_type(4)));
typedef __attribute__((address_space(1))) v4i* gptr_v4i;   // global (AS1)
typedef __attribute__((address_space(3))) v4i* lptr_v4i;   // LDS    (AS3)
#endif

// ---- helpers ---------------------------------------------------------------

__device__ __forceinline__ int f32_key(float f) {
  // monotonic float -> int mapping so atomicMax(int) == float max
  int i = __float_as_int(f);
  return (i >= 0) ? i : (i ^ 0x7fffffff);
}
__device__ __forceinline__ float key_f32(int k) {
  return __int_as_float((k >= 0) ? k : (k ^ 0x7fffffff));
}

// ---- conversion ------------------------------------------------------------

__global__ void k_f32_to_f16(const float* __restrict__ in,
                             _Float16* __restrict__ out, int n) {
  int i = blockIdx.x * blockDim.x + threadIdx.x;
  if (i < n) out[i] = (_Float16)in[i];
}

__global__ void k_fill_int(int* __restrict__ p, int v, int n) {
  int i = blockIdx.x * blockDim.x + threadIdx.x;
  if (i < n) p[i] = v;
}

// ---- WMMA GEMM: D[N,F] = A[N,K] (f16) * W[F,K]^T (f16), f32 accum ---------
// Block = TF waves; all waves share one 16-row A panel (tile row tm) staged in
// LDS via async global->LDS copies; wave w computes output tile (tm, tfBase+w).
// K stepped by 32 through v_wmma_f32_16x16x32_f16. Fragment layouts per ISA
// 7.12.2 (wave32):
//   A 16x32 f16: lane row=L%16, half=L/16 -> 8 f16 at k=half*8, 8 at 16+half*8
//   B 32x16 f16: lane col=L%16, half=L/16 -> 8 f16 at k=half*8, 8 at 16+half*8
//   D 16x16 f32: lane col=L%16; VGPR r -> row = r + 8*(L/16)

__global__ void k_wmma_gemm(const _Float16* __restrict__ A,
                            const _Float16* __restrict__ W,
                            float* __restrict__ D,
                            int K, int F, int TF) {
  int tfBlocks = (F >> 4) / TF;
  int tm = blockIdx.x / tfBlocks;
  int tf = (blockIdx.x % tfBlocks) * TF + (threadIdx.x >> 5);
  int lane = threadIdx.x & 31;
  int r16  = lane & 15;
  int half = lane >> 4;

  // ---- stage 16 x K A-panel (rows are contiguous: flat 16*K f16 copy) ----
  __shared__ _Float16 sA[16 * 256];
  const _Float16* gsrc = A + (size_t)tm * 16 * K;
  int chunks = (16 * K) >> 3;                       // 16-byte (8 x f16) chunks
#if USE_ASYNC_LDS
  for (int i = threadIdx.x; i < chunks; i += blockDim.x) {
    __builtin_amdgcn_global_load_async_to_lds_b128(
        (gptr_v4i)(gsrc + (size_t)i * 8),
        (lptr_v4i)(sA + i * 8),
        0, 0);
  }
  __builtin_amdgcn_s_wait_asynccnt(0);
#else
  for (int i = threadIdx.x; i < chunks; i += blockDim.x)
    ((uint4*)sA)[i] = ((const uint4*)gsrc)[i];
#endif
  __syncthreads();

  const _Float16* Ar = sA + r16 * K;                      // node row (LDS)
  const _Float16* Wr = W + (size_t)(tf * 16 + r16) * K;   // weight row (=B col)

  v8f acc = {};
  for (int kk = 0; kk < K; kk += 32) {
    v8h alo = *(const v8h*)(Ar + kk + half * 8);
    v8h ahi = *(const v8h*)(Ar + kk + 16 + half * 8);
    v8h blo = *(const v8h*)(Wr + kk + half * 8);
    v8h bhi = *(const v8h*)(Wr + kk + 16 + half * 8);
    v16h a, b;
#pragma unroll
    for (int i = 0; i < 8; ++i) {
      a[i] = alo[i]; a[i + 8] = ahi[i];
      b[i] = blo[i]; b[i + 8] = bhi[i];
    }
    acc = __builtin_amdgcn_wmma_f32_16x16x32_f16(
        /*neg_a=*/false, a, /*neg_b=*/false, b,
        /*c_mod=*/(short)0, acc, /*reuse_a=*/false, /*reuse_b=*/false);
  }
#pragma unroll
  for (int r = 0; r < 8; ++r) {
    int row = r + half * 8;
    D[(size_t)(tm * 16 + row) * F + tf * 16 + r16] = acc[r];
  }
}

// ---- attention scores: ss/sd[n,h] = <h[n,h,:], a_{s,d}[h,:]> ---------------

__global__ void k_scores(const float* __restrict__ hf,
                         const float* __restrict__ as,
                         const float* __restrict__ ad,
                         float* __restrict__ ss, float* __restrict__ sd,
                         int N, int H) {
  int idx = blockIdx.x * blockDim.x + threadIdx.x;
  if (idx >= N * H) return;
  int n = idx / H, hd = idx % H;
  const float* hp = hf + (size_t)n * H * 64 + hd * 64;
  const float* ap = as + hd * 64;
  const float* bp = ad + hd * 64;
  float s1 = 0.f, s2 = 0.f;
#pragma unroll 8
  for (int c = 0; c < 64; ++c) { float v = hp[c]; s1 += v * ap[c]; s2 += v * bp[c]; }
  ss[idx] = s1; sd[idx] = s2;
}

// ---- edge pass 1: per-dst segment max of leaky_relu(ss[src]+sd[dst]) ------

__global__ void k_edge_max(const int* __restrict__ ei, int E, int N,
                           const float* __restrict__ ss,
                           const float* __restrict__ sd,
                           int* __restrict__ mkey, int H) {
  int gid = blockIdx.x * blockDim.x + threadIdx.x;
  int ET = E + N;
  if (gid >= ET * H) return;
  int e = gid / H, hd = gid % H;
  int s, d;
  if (e < E) { s = ei[e]; d = ei[E + e]; } else { s = d = e - E; }
  float v = ss[s * H + hd] + sd[d * H + hd];
  v = (v > 0.f) ? v : 0.2f * v;                 // leaky_relu slope 0.2
  atomicMax(&mkey[d * H + hd], f32_key(v));
}

// ---- edge pass 2: per-(edge,head) weight w = exp(e - max), segsum += w -----

__global__ void k_edge_weight(const int* __restrict__ ei, int E, int N,
                              const float* __restrict__ ss,
                              const float* __restrict__ sd,
                              const int* __restrict__ mkey,
                              float* __restrict__ ew,
                              float* __restrict__ segsum, int H) {
  int gid = blockIdx.x * blockDim.x + threadIdx.x;
  int ET = E + N;
  if (gid >= ET * H) return;
  int e = gid / H, hd = gid % H;
  int s, d;
  if (e < E) { s = ei[e]; d = ei[E + e]; } else { s = d = e - E; }
  float v = ss[s * H + hd] + sd[d * H + hd];
  v = (v > 0.f) ? v : 0.2f * v;
  float w = __expf(v - key_f32(mkey[d * H + hd]));
  ew[gid] = w;
  atomicAdd(&segsum[d * H + hd], w);
}

// ---- edge pass 3: agg[dst] += h[src] * w ; 64 threads per edge -------------

__global__ void k_edge_agg(const int* __restrict__ ei, int E, int N,
                           const float* __restrict__ ew,
                           const float* __restrict__ hf,
                           float* __restrict__ agg, int H) {
  int gid = blockIdx.x * blockDim.x + threadIdx.x;
  int e = gid >> 6;
  int c = gid & 63;
  int ET = E + N;
  if (e >= ET) return;
  int s, d;
  if (e < E) { s = ei[e]; d = ei[E + e]; } else { s = d = e - E; }
  float w[4];
  for (int hd = 0; hd < H; ++hd) w[hd] = ew[e * H + hd];
  int F = H * 64;
  size_t sb = (size_t)s * F, db = (size_t)d * F;
  for (int hd = 0; hd < H; ++hd)
    atomicAdd(&agg[db + hd * 64 + c], hf[sb + hd * 64 + c] * w[hd]);
}

// ---- finalize: alpha-normalize + bias + (ELU) + LayerNorm ------------------
// One block per node, blockDim = F (= H*64, power of two <= 256).

__global__ void k_finalize_ln(const float* __restrict__ agg,
                              const float* __restrict__ segsum,
                              const float* __restrict__ bias,
                              const float* __restrict__ lnw,
                              const float* __restrict__ lnb,
                              float* __restrict__ out,
                              int H, int doElu) {
  int n = blockIdx.x;
  int F = blockDim.x;
  int j = threadIdx.x;
  int hd = j >> 6;
  float v = agg[(size_t)n * F + j] / (segsum[n * H + hd] + 1e-16f) + bias[j];
  if (doElu) v = (v > 0.f) ? v : (__expf(v) - 1.0f);

  __shared__ float red[256];
  float invF = 1.0f / (float)F;
  red[j] = v; __syncthreads();
  for (int st = F >> 1; st > 0; st >>= 1) {
    if (j < st) red[j] += red[j + st];
    __syncthreads();
  }
  float mu = red[0] * invF;
  __syncthreads();
  float dv = v - mu;
  red[j] = dv * dv; __syncthreads();
  for (int st = F >> 1; st > 0; st >>= 1) {
    if (j < st) red[j] += red[j + st];
    __syncthreads();
  }
  float var = red[0] * invF;
  out[(size_t)n * F + j] = dv * rsqrtf(var + 1e-5f) * lnw[j] + lnb[j];
}

// ---- MLP head pieces -------------------------------------------------------

__global__ void k_concat_f16(const float* __restrict__ h,
                             const float* __restrict__ role_table,
                             const int* __restrict__ role_ids,
                             _Float16* __restrict__ out, int N) {
  int i = blockIdx.x * blockDim.x + threadIdx.x;
  if (i >= N * 128) return;
  int n = i >> 7, j = i & 127;
  float v = (j < 64) ? h[(size_t)n * 64 + j]
                     : role_table[role_ids[n] * 64 + (j - 64)];
  out[i] = (_Float16)v;
}

__global__ void k_relu_bias_f16(const float* __restrict__ z,
                                const float* __restrict__ b,
                                _Float16* __restrict__ out, int total, int F) {
  int i = blockIdx.x * blockDim.x + threadIdx.x;
  if (i >= total) return;
  float t = z[i] + b[i % F];
  out[i] = (_Float16)((t > 0.f) ? t : 0.f);
}

__global__ void k_bias_out(const float* __restrict__ z,
                           const float* __restrict__ b,
                           float* __restrict__ out, int total, int F) {
  int i = blockIdx.x * blockDim.x + threadIdx.x;
  if (i >= total) return;
  out[i] = z[i] + b[i % F];
}

// ---------------------------------------------------------------------------

extern "C" void kernel_launch(void* const* d_in, const int* in_sizes, int n_in,
                              void* d_out, int out_size, void* d_ws, size_t ws_size,
                              hipStream_t stream) {
  const float* x          = (const float*)d_in[0];
  const int*   ei         = (const int*)d_in[1];
  const int*   role_ids   = (const int*)d_in[2];
  const float* W1         = (const float*)d_in[3];
  const float* a1s        = (const float*)d_in[4];
  const float* a1d        = (const float*)d_in[5];
  const float* b1         = (const float*)d_in[6];
  const float* W2         = (const float*)d_in[7];
  const float* a2s        = (const float*)d_in[8];
  const float* a2d        = (const float*)d_in[9];
  const float* b2         = (const float*)d_in[10];
  const float* W3         = (const float*)d_in[11];
  const float* a3s        = (const float*)d_in[12];
  const float* a3d        = (const float*)d_in[13];
  const float* b3         = (const float*)d_in[14];
  const float* ln1w       = (const float*)d_in[15];
  const float* ln1b       = (const float*)d_in[16];
  const float* ln2w       = (const float*)d_in[17];
  const float* ln2b       = (const float*)d_in[18];
  const float* ln3w       = (const float*)d_in[19];
  const float* ln3b       = (const float*)d_in[20];
  const float* role_table = (const float*)d_in[21];
  const float* p1w        = (const float*)d_in[22];
  const float* p1b        = (const float*)d_in[23];
  const float* p2w        = (const float*)d_in[24];
  const float* p2b        = (const float*)d_in[25];
  (void)n_in; (void)out_size; (void)ws_size;

  const int E = in_sizes[1] / 2;
  const int N = in_sizes[2];
  const int ET = E + N;

  // ---- carve workspace ----
  char* base = (char*)d_ws;
  size_t off = 0;
  auto carve = [&](size_t bytes) -> void* {
    void* p = base + off;
    off += (bytes + 255) & ~(size_t)255;
    return p;
  };
  _Float16* h16    = (_Float16*)carve((size_t)N * 256 * sizeof(_Float16));
  float*    featA  = (float*)carve((size_t)N * 256 * sizeof(float));
  float*    featB  = (float*)carve((size_t)N * 256 * sizeof(float));
  float*    agg    = (float*)carve((size_t)N * 256 * sizeof(float));
  float*    ss     = (float*)carve((size_t)N * 4 * sizeof(float));
  float*    sd     = (float*)carve((size_t)N * 4 * sizeof(float));
  int*      mkey   = (int*)carve((size_t)N * 4 * sizeof(int));
  float*    segsum = (float*)carve((size_t)N * 4 * sizeof(float));
  float*    ew     = (float*)carve((size_t)ET * 4 * sizeof(float));
  _Float16* W1h    = (_Float16*)carve(256 * 64 * sizeof(_Float16));
  _Float16* W2h    = (_Float16*)carve(256 * 256 * sizeof(_Float16));
  _Float16* W3h    = (_Float16*)carve(64 * 256 * sizeof(_Float16));
  _Float16* p1h    = (_Float16*)carve(64 * 128 * sizeof(_Float16));
  _Float16* p2h    = (_Float16*)carve(64 * 64 * sizeof(_Float16));

  auto cvt = [&](const float* src, _Float16* dst, int n) {
    k_f32_to_f16<<<(n + 255) / 256, 256, 0, stream>>>(src, dst, n);
  };
  auto gemm = [&](const _Float16* A, const _Float16* Wt, float* Dm, int K, int F) {
    int tilesF = F / 16;
    int TF = tilesF < 8 ? tilesF : 8;          // waves per block
    int tfBlocks = tilesF / TF;
    int blocks = (N / 16) * tfBlocks;
    k_wmma_gemm<<<blocks, TF * 32, 0, stream>>>(A, Wt, Dm, K, F, TF);
  };
  auto edge_phase = [&](const float* hf, const float* as, const float* ad, int H) {
    k_scores<<<(N * H + 255) / 256, 256, 0, stream>>>(hf, as, ad, ss, sd, N, H);
    k_fill_int<<<(N * H + 255) / 256, 256, 0, stream>>>(mkey, (int)0x80000000, N * H);
    (void)hipMemsetAsync(segsum, 0, (size_t)N * H * sizeof(float), stream);
    (void)hipMemsetAsync(agg, 0, (size_t)N * H * 64 * sizeof(float), stream);
    k_edge_max<<<(ET * H + 255) / 256, 256, 0, stream>>>(ei, E, N, ss, sd, mkey, H);
    k_edge_weight<<<(ET * H + 255) / 256, 256, 0, stream>>>(ei, E, N, ss, sd, mkey,
                                                            ew, segsum, H);
    k_edge_agg<<<(ET * 64 + 255) / 256, 256, 0, stream>>>(ei, E, N, ew, hf, agg, H);
  };

  // ---- convert weights to f16 once per call ----
  cvt(W1, W1h, 256 * 64);
  cvt(W2, W2h, 256 * 256);
  cvt(W3, W3h, 64 * 256);
  cvt(p1w, p1h, 64 * 128);
  cvt(p2w, p2h, 64 * 64);

  // ---- GAT layer 1: x[N,64] -> featA[N,256] ----
  cvt(x, h16, N * 64);
  gemm(h16, W1h, featB, 64, 256);
  edge_phase(featB, a1s, a1d, 4);
  k_finalize_ln<<<N, 256, 0, stream>>>(agg, segsum, b1, ln1w, ln1b, featA, 4, /*elu=*/1);

  // ---- GAT layer 2: featA[N,256] -> featA[N,256] ----
  cvt(featA, h16, N * 256);
  gemm(h16, W2h, featB, 256, 256);
  edge_phase(featB, a2s, a2d, 4);
  k_finalize_ln<<<N, 256, 0, stream>>>(agg, segsum, b2, ln2w, ln2b, featA, 4, /*elu=*/1);

  // ---- GAT layer 3: featA[N,256] -> featA[N,64] (1 head, no ELU) ----
  cvt(featA, h16, N * 256);
  gemm(h16, W3h, featB, 256, 64);
  edge_phase(featB, a3s, a3d, 1);
  k_finalize_ln<<<N, 64, 0, stream>>>(agg, segsum, b3, ln3w, ln3b, featA, 1, /*elu=*/0);

  // ---- MLP head: concat -> relu(p1) -> p2 ----
  k_concat_f16<<<(N * 128 + 255) / 256, 256, 0, stream>>>(featA, role_table, role_ids, h16, N);
  gemm(h16, p1h, featB, 128, 64);
  k_relu_bias_f16<<<(N * 64 + 255) / 256, 256, 0, stream>>>(featB, p1b, h16, N * 64, 64);
  gemm(h16, p2h, featB, 64, 64);
  k_bias_out<<<(N * 64 + 255) / 256, 256, 0, stream>>>(featB, p2b, (float*)d_out, N * 64, 64);
}
